// PrototypeClassifier_56495999812266
// MI455X (gfx1250) — matrix operand order repaired and checked by reading, not compiled, not run on hardware
//
#include <hip/hip_runtime.h>

// Cosine similarity: out[n][c] = <x_n, p_c> / (max(|x_n|,eps)*max(|p_c|,eps))
// x: [16384, 256] f32, prototypes: [10000, 256] f32, out: [16384, 10000] f32.
// GEMM in bf16 WMMA (16x16x32, f32 accum), norms in f32.

#define NROWS 16384
#define CCOLS 10000
#define DIMK  256
#define EPSN  1e-6f

typedef __attribute__((ext_vector_type(16))) __bf16 v16bf;
typedef __attribute__((ext_vector_type(8)))  float  v8f;

union FragU {
    uint4 q[2];
    v16bf v;
};

union PackU {
    __bf16 h[4];
    uint2  u;
};

// ---------------------------------------------------------------------------
// Reciprocal L2 norms: one wave32 per row. rx[NROWS], rp[CCOLS] in workspace.
// ---------------------------------------------------------------------------
__global__ __launch_bounds__(256)
void norms_kernel(const float* __restrict__ X, const float* __restrict__ P,
                  float* __restrict__ rx, float* __restrict__ rp) {
    int wave = blockIdx.x * (blockDim.x >> 5) + (threadIdx.x >> 5);
    int l = threadIdx.x & 31;
    const float* src;
    float* dst;
    if (wave < NROWS) {
        src = X + (size_t)wave * DIMK;
        dst = rx + wave;
    } else {
        int pr = wave - NROWS;
        if (pr >= CCOLS) return;
        src = P + (size_t)pr * DIMK;
        dst = rp + pr;
    }
    float s = 0.0f;
#pragma unroll
    for (int i = 0; i < DIMK / 32; ++i) {
        float v = src[l + i * 32];
        s += v * v;
    }
#pragma unroll
    for (int off = 16; off >= 1; off >>= 1)
        s += __shfl_xor(s, off, 32);
    if (l == 0) *dst = 1.0f / fmaxf(sqrtf(s), EPSN);
}

// ---------------------------------------------------------------------------
// Tiled bf16 WMMA GEMM with fp32 normalization epilogue.
// Block: 256 threads (8 waves). Output tile per block: 128 x 128.
// Per wave: 32 rows x 64 cols = 2x4 grid of 16x16 accumulators.
// ---------------------------------------------------------------------------
#define LDS_STRIDE 40   // ushorts per 32-element K row (80B; conflict-free)

__global__ __launch_bounds__(256)
void cos_gemm(const float* __restrict__ X, const float* __restrict__ P,
              const float* __restrict__ rx, const float* __restrict__ rp,
              float* __restrict__ out) {
    __shared__ unsigned short As[128 * LDS_STRIDE];
    __shared__ unsigned short Bs[128 * LDS_STRIDE];

    const int t = threadIdx.x;
    const int l = t & 31;
    const int w = t >> 5;           // wave id 0..7
    const int wr = w & 3;           // 4 row groups of 32
    const int wc = w >> 2;          // 2 col groups of 64
    const int fr = l & 15;          // fragment row within 16
    const int kh = l >> 4;          // K half selector

    const int growBase = blockIdx.y * 128;   // x-row base
    const int gcolBase = blockIdx.x * 128;   // prototype base

    v8f acc[2][4];
#pragma unroll
    for (int m = 0; m < 2; ++m)
#pragma unroll
        for (int n = 0; n < 4; ++n)
#pragma unroll
            for (int i = 0; i < 8; ++i)
                acc[m][n][i] = 0.0f;

    // Per-thread staging coordinates: 128 rows x 8 float4 per 32-wide K slab.
    const int ldRow = t >> 3;       // 0..31  (+i*32 per chunk)
    const int ldC4  = t & 7;        // float4 column within row

    for (int kt = 0; kt < DIMK / 32; ++kt) {
        const int kBase = kt * 32;

        // ---- global loads: 128x32 f32 of A and B (4 float4 per thread each)
        float4 ga[4], gb[4];
#pragma unroll
        for (int i = 0; i < 4; ++i) {
            int row = ldRow + i * 32;
            ga[i] = *(const float4*)(X + (size_t)(growBase + row) * DIMK + kBase + ldC4 * 4);
            // Rows >= CCOLS only feed output columns that the epilogue guard
            // discards, so clamping (instead of zero-fill) is safe and avoids
            // exec-mask branches.
            int prow = gcolBase + row;
            prow = prow < CCOLS ? prow : (CCOLS - 1);
            gb[i] = *(const float4*)(P + (size_t)prow * DIMK + kBase + ldC4 * 4);
        }

        __syncthreads();   // previous iteration's LDS reads complete

        // ---- convert to bf16 (native fptrunc -> packed cvt), stage into LDS
#pragma unroll
        for (int i = 0; i < 4; ++i) {
            int row = ldRow + i * 32;
            PackU ua, ub;
            ua.h[0] = (__bf16)ga[i].x; ua.h[1] = (__bf16)ga[i].y;
            ua.h[2] = (__bf16)ga[i].z; ua.h[3] = (__bf16)ga[i].w;
            ub.h[0] = (__bf16)gb[i].x; ub.h[1] = (__bf16)gb[i].y;
            ub.h[2] = (__bf16)gb[i].z; ub.h[3] = (__bf16)gb[i].w;
            *(uint2*)&As[row * LDS_STRIDE + ldC4 * 4] = ua.u;
            *(uint2*)&Bs[row * LDS_STRIDE + ldC4 * 4] = ub.u;
        }

        __syncthreads();

        // ---- fragment loads per ISA VGPR layouts
        // A 16x32 bf16: lane needs K = kh*8..kh*8+7 and 16+kh*8..16+kh*8+7
        FragU a[2];
#pragma unroll
        for (int m = 0; m < 2; ++m) {
            int ra = (wr * 32 + m * 16 + fr) * LDS_STRIDE + kh * 8;
            a[m].q[0] = *(const uint4*)&As[ra];
            a[m].q[1] = *(const uint4*)&As[ra + 16];
        }
        // B 32x16 bf16 (col N = lane&15): lane needs K = kh*8..kh*8+7 and 16+kh*8..16+kh*8+7
        FragU b[4];
#pragma unroll
        for (int n = 0; n < 4; ++n) {
            int rb = (wc * 64 + n * 16 + fr) * LDS_STRIDE + kh * 8;
            b[n].q[0] = *(const uint4*)&Bs[rb];
            b[n].q[1] = *(const uint4*)&Bs[rb + 16];
        }

        // ---- 8 WMMAs per K-step
#pragma unroll
        for (int m = 0; m < 2; ++m)
#pragma unroll
            for (int n = 0; n < 4; ++n)
                acc[m][n] = __builtin_amdgcn_wmma_f32_16x16x32_bf16(
                    /*neg_a=*/false, a[m].v, /*neg_b=*/false, b[n].v,
                    /*c_mod=*/(short)0, acc[m][n],
                    /*reuse_a=*/false, /*reuse_b=*/false);
    }

    // ---- epilogue: scale by reciprocal norms, guarded non-temporal store
    // C/D layout: lanes 0-15: VGPR r -> M=r, N=lane ; lanes 16-31: M=r+8, N=lane-16
#pragma unroll
    for (int m = 0; m < 2; ++m) {
        const int rowBase = growBase + wr * 32 + m * 16 + kh * 8;
        float rxv[8];
#pragma unroll
        for (int r = 0; r < 8; ++r)
            rxv[r] = rx[rowBase + r];
#pragma unroll
        for (int n = 0; n < 4; ++n) {
            int col = gcolBase + wc * 64 + n * 16 + fr;
            if (col < CCOLS) {
                float sp = rp[col];
#pragma unroll
                for (int r = 0; r < 8; ++r) {
                    __builtin_nontemporal_store(acc[m][n][r] * rxv[r] * sp,
                                                &out[(size_t)(rowBase + r) * CCOLS + col]);
                }
            }
        }
    }
}

// ---------------------------------------------------------------------------
extern "C" void kernel_launch(void* const* d_in, const int* in_sizes, int n_in,
                              void* d_out, int out_size, void* d_ws, size_t ws_size,
                              hipStream_t stream) {
    (void)in_sizes; (void)n_in; (void)out_size; (void)ws_size;
    const float* X = (const float*)d_in[0];
    const float* P = (const float*)d_in[1];
    float* rx = (float*)d_ws;          // [NROWS]
    float* rp = rx + NROWS;            // [CCOLS]
    float* out = (float*)d_out;

    int totalWaves = NROWS + CCOLS;    // one wave per row
    int nb = (totalWaves + 7) / 8;     // 8 waves per 256-thread block
    norms_kernel<<<nb, 256, 0, stream>>>(X, P, rx, rp);

    dim3 grid((CCOLS + 127) / 128, NROWS / 128);  // 79 x 128 tiles
    cos_gemm<<<grid, 256, 0, stream>>>(X, P, rx, rp, out);
}
